// LatentDiffusionReasoner_9088150798901
// MI455X (gfx1250) — compile-verified, matrix operand order checked by reading, not run on hardware
//
#include <hip/hip_runtime.h>
#include <hip/hip_bf16.h>
#include <math.h>
#include <stdint.h>

// LatentDiffusionReasoner for MI455X (gfx1250, wave32, WMMA + TDM).
// B=64, S=64, D=1024, H=16, HD=64, FFN=2048, steps=8, T_EMB=9, CLIP=1.0

#define D_MODEL 1024
#define NTOK    4096            // B*S
#define D_FFN   2048
#define LN_EPS  1e-5f
#define CLIPV   1.0f

// TDM writes LDS behind the compiler's back (the builtin carries descriptor
// VALUES, no pointer). Escaping the LDS pointer into a may-write asm breaks
// the "LDS global is never stored -> loads are undef" proof, so the ds_load
// fragment reads must be materialized.
__device__ __forceinline__ void tdm_lds_publish(const void* lds_ptr) {
  asm volatile("" ::"v"(lds_ptr) : "memory");
}

typedef __attribute__((ext_vector_type(16))) __bf16 v16bf;
typedef __attribute__((ext_vector_type(8)))  float  v8f;
typedef __attribute__((ext_vector_type(4)))  unsigned int u32x4;
typedef __attribute__((ext_vector_type(8)))  int i32x8;
typedef __attribute__((ext_vector_type(4)))  int i32x4;

union FragBF {
  v16bf  v;
  float4 q[2];   // two 16B chunks per lane
};

// ---------------------------------------------------------------- reductions
__device__ __forceinline__ float block_reduce_sum(float v, float* red) {
  const int tid = threadIdx.x;
  red[tid] = v;
  __syncthreads();
  for (int off = 128; off > 0; off >>= 1) {
    if (tid < off) red[tid] += red[tid + off];
    __syncthreads();
  }
  float r = red[0];
  __syncthreads();
  return r;
}

// ------------------------------------------------------------- TDM 2D load
// Issue TENSOR_LOAD_TO_LDS of a (tile_rows x 32) bf16 tile whose rows are
// row_elems apart in global memory, into LDS at lds_byte_off with 16B padding
// after every 64B row (LDS row stride = 40 bf16 = 80B).
// D# packing per CDNA5 ISA (group0 128b, group1 256b).
// This toolchain uses the 6-arg builtin: (g0, g1, g2, g3, g4, cpol).
__device__ __forceinline__ void tdm_load_tile(
    unsigned lds_byte_off, const __bf16* gptr, int row_elems, int tile_rows) {
  const unsigned long long ga = (unsigned long long)(size_t)gptr;
  u32x4 g0;
  g0[0] = 1u;                                              // count=1, user mode
  g0[1] = lds_byte_off;                                    // lds_addr
  g0[2] = (unsigned)(ga & 0xffffffffu);                    // global_addr[31:0]
  g0[3] = (unsigned)((ga >> 32) & 0x01ffffffu) | (2u << 30); // addr[56:32], type=2
  i32x8 g1;
  // data_size=2B (1<<16), pad_enable (1<<20), pad_interval=64B (3<<22),
  // pad_amount=16B (3<<25)
  g1[0] = (1 << 16) | (1 << 20) | (3 << 22) | (3 << 25);
  g1[1] = (row_elems & 0xffff) << 16;                      // tensor_dim0[15:0]
  g1[2] = ((row_elems >> 16) & 0xffff);                    // tensor_dim0[31:16]
                                                           // tensor_dim1[15:0]=0
  g1[3] = 16 | (32 << 16);                                 // tensor_dim1=1<<20; tile_dim0=32
  g1[4] = tile_rows & 0xffff;                              // tile_dim1; tile_dim2=0
  g1[5] = row_elems;                                       // tensor_dim0_stride[31:0]
  g1[6] = 0;                                               // stride hi, dim1_stride lo
  g1[7] = 0;
  const i32x4 z4 = {};
  const i32x8 z8 = {};
  __builtin_amdgcn_tensor_load_to_lds(g0, g1, z4, z4, z8, 0);
}

// ------------------------------------------------------------------- LN (+add)
__global__ __launch_bounds__(256) void ln_kernel(
    const float* __restrict__ h, const float* __restrict__ add,
    const float* __restrict__ temb, const float* __restrict__ g,
    const float* __restrict__ bias, __bf16* __restrict__ out) {
  __shared__ float red[256];
  const int token = blockIdx.x;
  const int tid = threadIdx.x;
  const float* hrow = h + (size_t)token * D_MODEL;
  const float* arow = add ? add + (size_t)token * D_MODEL : nullptr;
  float x[4];
  float s = 0.f;
#pragma unroll
  for (int i = 0; i < 4; ++i) {
    const int idx = tid + i * 256;
    float v = hrow[idx] + temb[idx];
    if (arow) v += arow[idx];
    x[i] = v;
    s += v;
  }
  const float mean = block_reduce_sum(s, red) * (1.f / D_MODEL);
  float s2 = 0.f;
#pragma unroll
  for (int i = 0; i < 4; ++i) { const float d = x[i] - mean; s2 += d * d; }
  const float var = block_reduce_sum(s2, red) * (1.f / D_MODEL);
  const float rstd = rsqrtf(var + LN_EPS);
  __bf16* orow = out + (size_t)token * D_MODEL;
#pragma unroll
  for (int i = 0; i < 4; ++i) {
    const int idx = tid + i * 256;
    orow[idx] = (__bf16)((x[i] - mean) * rstd * g[idx] + bias[idx]);
  }
}

// --------------------------------------------------------------- f32 -> bf16
__global__ __launch_bounds__(256) void f32_to_bf16_kernel(
    const float* __restrict__ in, __bf16* __restrict__ out, int n) {
  const int i = blockIdx.x * 256 + threadIdx.x;
  if (i < n) out[i] = (__bf16)in[i];
}

// --------------------------------------------------------- WMMA GEMM (bf16)
// C[M,N] = X[M,K] @ W[N,K]^T. 256 threads = 8 waves; block tile 128x64.
// TDM double-buffers A(128x32) and B(64x32) tiles into LDS while waves
// run 4 back-to-back WMMAs per K-chunk.
#define GM_F32       0
#define GM_BF16      1
#define GM_SILU_BF16 2
#define GM_BIAS_F32  3

#define ABUF (128 * 40)          // elements, row stride 40 (TDM-padded)
#define BBUF (64 * 40)
#define BUFSZ (ABUF + BBUF)      // 7680 elements = 15360 bytes per buffer

__global__ __launch_bounds__(256) void gemm_bf16_wmma_kernel(
    const __bf16* __restrict__ X, const __bf16* __restrict__ W,
    const float* __restrict__ bias, float* __restrict__ outF,
    __bf16* __restrict__ outB, int K, int N, int mode) {
  __shared__ __bf16 smem[2 * BUFSZ];  // [buf0: A,B][buf1: A,B]
  const int m0 = blockIdx.x * 128;
  const int n0 = blockIdx.y * 64;
  const int tid  = threadIdx.x;
  const int wave = tid >> 5;
  const int lane = tid & 31;
  const int fr = lane & 15;          // row (A) / col (B) within 16-tile
  const int fk = (lane >> 4) * 8;    // K interleave half (ISA 16-bit A/B layout)

  const __bf16* Xt = X + (size_t)m0 * K;
  const __bf16* Wt = W + (size_t)n0 * K;
  const int nchunks = K >> 5;

  if (wave == 0) {  // TDM prologue: fill buffer 0 (2 descriptors in flight)
    tdm_load_tile(0u, Xt, K, 128);
    tdm_load_tile((unsigned)(ABUF * 2), Wt, K, 64);
  }
  tdm_lds_publish(smem);

  v8f acc[4] = {};
  for (int i = 0; i < nchunks; ++i) {
    const int cur = i & 1;
    if (wave == 0) {
      if (i + 1 < nchunks) {   // prefetch next chunk into other buffer
        const unsigned nb = (unsigned)(((i + 1) & 1) * BUFSZ * 2);
        tdm_load_tile(nb, Xt + (i + 1) * 32, K, 128);
        tdm_load_tile(nb + ABUF * 2, Wt + (i + 1) * 32, K, 64);
        __builtin_amdgcn_s_wait_tensorcnt(2);  // current buffer done
      } else {
        __builtin_amdgcn_s_wait_tensorcnt(0);
      }
    }
    tdm_lds_publish(smem);  // TDM wrote LDS: fragment loads must hit memory
    __syncthreads();        // publish LDS tiles to all 8 waves

    const __bf16* As = &smem[cur * BUFSZ];
    const __bf16* Bs = &smem[cur * BUFSZ + ABUF];
    FragBF a, b[4];
    const __bf16* ap = &As[(wave * 16 + fr) * 40 + fk];
    a.q[0] = *(const float4*)ap;
    a.q[1] = *(const float4*)(ap + 16);
#pragma unroll
    for (int nt = 0; nt < 4; ++nt) {   // batch all B-frag loads
      const __bf16* bp = &Bs[(nt * 16 + fr) * 40 + fk];
      b[nt].q[0] = *(const float4*)bp;
      b[nt].q[1] = *(const float4*)(bp + 16);
    }
#pragma unroll
    for (int nt = 0; nt < 4; ++nt)     // 4 WMMAs issue back-to-back
      acc[nt] = __builtin_amdgcn_wmma_f32_16x16x32_bf16(
          false, a.v, false, b[nt].v, (short)0, acc[nt], false, false);
    __syncthreads();  // all waves done reading before TDM overwrites this buf
  }
  // epilogue: C/D layout — VGPR j: lanes0-15 M=j, lanes16-31 M=j+8; N=lane&15
#pragma unroll
  for (int nt = 0; nt < 4; ++nt) {
#pragma unroll
    for (int j = 0; j < 8; ++j) {
      const int row = m0 + wave * 16 + j + ((lane >> 4) << 3);
      const int col = n0 + nt * 16 + fr;
      float v = acc[nt][j];
      if (bias) v += bias[col];
      if (mode == GM_SILU_BF16) v = v / (1.f + __expf(-v));
      if (mode == GM_F32 || mode == GM_BIAS_F32)
        outF[(size_t)row * N + col] = v;
      else
        outB[(size_t)row * N + col] = (__bf16)v;
    }
  }
}

// ------------------------------------------- V transpose: [tok,chan]->[b,h,d,s]
__global__ __launch_bounds__(256) void transpose_v_kernel(
    const __bf16* __restrict__ v, __bf16* __restrict__ vT) {
  const int i = blockIdx.x * 256 + threadIdx.x;   // over NTOK*D
  const int token = i >> 10;
  const int c = i & 1023;
  const int b = token >> 6, s = token & 63;
  const int hh = c >> 6, d = c & 63;
  vT[(size_t)((((b << 4) + hh) << 6) + d) * 64 + s] = v[i];
}

// ---------------------------------------------------------- attention per (b,h)
__global__ __launch_bounds__(256) void attention_kernel(
    const __bf16* __restrict__ q, const __bf16* __restrict__ k,
    const __bf16* __restrict__ vT, __bf16* __restrict__ attn_out) {
  __shared__ __bf16 qs[64 * 72];
  __shared__ __bf16 ks[64 * 72];
  __shared__ __bf16 vs[64 * 72];
  __shared__ __bf16 ps[64 * 72];
  __shared__ float  sc[64 * 68];
  const int bh = blockIdx.x;
  const int b = bh >> 4, hh = bh & 15;
  const int tid = threadIdx.x;
  const __bf16* qbase = q + (size_t)(b * 64) * D_MODEL + hh * 64;
  const __bf16* kbase = k + (size_t)(b * 64) * D_MODEL + hh * 64;
  const __bf16* vbase = vT + (size_t)bh * 64 * 64;
  {  // stage 64x64 q, k, vT tiles
    const int row = tid >> 2, cb = (tid & 3) * 16;
    *(float4*)&qs[row * 72 + cb]     = *(const float4*)&qbase[row * D_MODEL + cb];
    *(float4*)&qs[row * 72 + cb + 8] = *(const float4*)&qbase[row * D_MODEL + cb + 8];
    *(float4*)&ks[row * 72 + cb]     = *(const float4*)&kbase[row * D_MODEL + cb];
    *(float4*)&ks[row * 72 + cb + 8] = *(const float4*)&kbase[row * D_MODEL + cb + 8];
    *(float4*)&vs[row * 72 + cb]     = *(const float4*)&vbase[row * 64 + cb];
    *(float4*)&vs[row * 72 + cb + 8] = *(const float4*)&vbase[row * 64 + cb + 8];
  }
  __syncthreads();

  const int wave = tid >> 5, lane = tid & 31;
  const int fr = lane & 15;
  const int fk = (lane >> 4) * 8;
  const int mt  = wave >> 1;        // query 16-tile
  const int nt0 = (wave & 1) * 2;   // first of two key/d 16-tiles

  // phase 1: scores = q k^T / 8
  v8f s0 = {}, s1 = {};
#pragma unroll
  for (int kc = 0; kc < 64; kc += 32) {
    FragBF a, b0, b1;
    const __bf16* ap = &qs[(mt * 16 + fr) * 72 + kc + fk];
    a.q[0] = *(const float4*)ap; a.q[1] = *(const float4*)(ap + 16);
    const __bf16* bp0 = &ks[(nt0 * 16 + fr) * 72 + kc + fk];
    b0.q[0] = *(const float4*)bp0; b0.q[1] = *(const float4*)(bp0 + 16);
    const __bf16* bp1 = &ks[((nt0 + 1) * 16 + fr) * 72 + kc + fk];
    b1.q[0] = *(const float4*)bp1; b1.q[1] = *(const float4*)(bp1 + 16);
    s0 = __builtin_amdgcn_wmma_f32_16x16x32_bf16(false, a.v, false, b0.v, (short)0, s0, false, false);
    s1 = __builtin_amdgcn_wmma_f32_16x16x32_bf16(false, a.v, false, b1.v, (short)0, s1, false, false);
  }
#pragma unroll
  for (int j = 0; j < 8; ++j) {
    const int r = mt * 16 + j + ((lane >> 4) << 3);
    sc[r * 68 + nt0 * 16 + fr]       = s0[j] * 0.125f;   // 1/sqrt(64)
    sc[r * 68 + (nt0 + 1) * 16 + fr] = s1[j] * 0.125f;
  }
  __syncthreads();

  // softmax: one thread per query row
  if (tid < 64) {
    float mx = -1e30f;
    for (int c = 0; c < 64; ++c) mx = fmaxf(mx, sc[tid * 68 + c]);
    float sum = 0.f;
    for (int c = 0; c < 64; ++c) {
      const float e = __expf(sc[tid * 68 + c] - mx);
      sc[tid * 68 + c] = e;
      sum += e;
    }
    const float inv = 1.f / sum;
    for (int c = 0; c < 64; ++c) ps[tid * 72 + c] = (__bf16)(sc[tid * 68 + c] * inv);
  }
  __syncthreads();

  // phase 2: out = probs @ v
  v8f o0 = {}, o1 = {};
#pragma unroll
  for (int kc = 0; kc < 64; kc += 32) {
    FragBF a, b0, b1;
    const __bf16* ap = &ps[(mt * 16 + fr) * 72 + kc + fk];
    a.q[0] = *(const float4*)ap; a.q[1] = *(const float4*)(ap + 16);
    const __bf16* bp0 = &vs[(nt0 * 16 + fr) * 72 + kc + fk];
    b0.q[0] = *(const float4*)bp0; b0.q[1] = *(const float4*)(bp0 + 16);
    const __bf16* bp1 = &vs[((nt0 + 1) * 16 + fr) * 72 + kc + fk];
    b1.q[0] = *(const float4*)bp1; b1.q[1] = *(const float4*)(bp1 + 16);
    o0 = __builtin_amdgcn_wmma_f32_16x16x32_bf16(false, a.v, false, b0.v, (short)0, o0, false, false);
    o1 = __builtin_amdgcn_wmma_f32_16x16x32_bf16(false, a.v, false, b1.v, (short)0, o1, false, false);
  }
  __bf16* obase = attn_out + (size_t)(b * 64) * D_MODEL + hh * 64;
#pragma unroll
  for (int j = 0; j < 8; ++j) {
    const int r = mt * 16 + j + ((lane >> 4) << 3);
    obase[r * D_MODEL + nt0 * 16 + fr]       = (__bf16)o0[j];
    obase[r * D_MODEL + (nt0 + 1) * 16 + fr] = (__bf16)o1[j];
  }
}

// ------------------------------------------------- norm-clip residual update
__global__ __launch_bounds__(256) void update_kernel(
    float* __restrict__ h, const float* __restrict__ a,
    const float* __restrict__ f, const int* __restrict__ steps_p) {
  __shared__ float red[256];
  const int token = blockIdx.x;
  const int tid = threadIdx.x;
  const size_t base = (size_t)token * D_MODEL;
  float u[4];
  float s2 = 0.f;
#pragma unroll
  for (int i = 0; i < 4; ++i) {
    const int idx = tid + i * 256;
    u[i] = a[base + idx] + f[base + idx];
    s2 += u[i] * u[i];
  }
  const float norm = sqrtf(block_reduce_sum(s2, red));
  const float stepsF = (float)(*steps_p);
  const float scale = fminf(CLIPV / fmaxf(norm, 1e-6f), 1.0f) / stepsF;
#pragma unroll
  for (int i = 0; i < 4; ++i) {
    const int idx = tid + i * 256;
    h[base + idx] += u[i] * scale;
  }
}

// ---------------------------------------------------------------------- host
extern "C" void kernel_launch(void* const* d_in, const int* in_sizes, int n_in,
                              void* d_out, int out_size, void* d_ws, size_t ws_size,
                              hipStream_t stream) {
  (void)in_sizes; (void)n_in; (void)out_size; (void)ws_size;
  const float* latents = (const float*)d_in[0];
  const float* t_table = (const float*)d_in[1];
  const float* attn_g  = (const float*)d_in[2];
  const float* attn_b  = (const float*)d_in[3];
  const float* ffn_g   = (const float*)d_in[4];
  const float* ffn_b   = (const float*)d_in[5];
  const float* Wq      = (const float*)d_in[6];
  const float* Wk      = (const float*)d_in[7];
  const float* Wv      = (const float*)d_in[8];
  const float* Wo      = (const float*)d_in[9];
  const float* Win     = (const float*)d_in[10];
  const float* b_in    = (const float*)d_in[11];
  const float* Wout    = (const float*)d_in[12];
  const float* b_out   = (const float*)d_in[13];
  const int*   steps_p = (const int*)d_in[14];

  const size_t DD  = (size_t)D_MODEL * D_MODEL;   // 1M
  const size_t D2D = 2 * DD;                      // 2M
  const size_t TD  = (size_t)NTOK * D_MODEL;      // 4M

  char* ws = (char*)d_ws;
  size_t off = 0;
  auto alloc_bf = [&](size_t elems) {
    __bf16* p = (__bf16*)(ws + off); off += elems * sizeof(__bf16); return p;
  };
  __bf16* Wq16   = alloc_bf(DD);
  __bf16* Wk16   = alloc_bf(DD);
  __bf16* Wv16   = alloc_bf(DD);
  __bf16* Wo16   = alloc_bf(DD);
  __bf16* Win16  = alloc_bf(D2D);
  __bf16* Wout16 = alloc_bf(D2D);
  __bf16* x16    = alloc_bf(TD);     // reused as ffn_in
  __bf16* q16    = alloc_bf(TD);
  __bf16* k16    = alloc_bf(TD);
  __bf16* v16    = alloc_bf(TD);
  __bf16* vT16   = alloc_bf(TD);
  __bf16* hid16  = v16;              // aliases v16+vT16 (both dead by FFN1)
  __bf16* attn16 = alloc_bf(TD);
  float* attn_upd = (float*)(ws + off); off += TD * sizeof(float);
  float* ffn_upd  = (float*)(ws + off); off += TD * sizeof(float);

  float* h = (float*)d_out;
  (void)hipMemcpyAsync(h, latents, TD * sizeof(float), hipMemcpyDeviceToDevice, stream);

  // one-time (per launch) weight conversion to bf16 — stays hot in 192MB L2
  f32_to_bf16_kernel<<<(int)(DD  / 256), 256, 0, stream>>>(Wq,   Wq16,   (int)DD);
  f32_to_bf16_kernel<<<(int)(DD  / 256), 256, 0, stream>>>(Wk,   Wk16,   (int)DD);
  f32_to_bf16_kernel<<<(int)(DD  / 256), 256, 0, stream>>>(Wv,   Wv16,   (int)DD);
  f32_to_bf16_kernel<<<(int)(DD  / 256), 256, 0, stream>>>(Wo,   Wo16,   (int)DD);
  f32_to_bf16_kernel<<<(int)(D2D / 256), 256, 0, stream>>>(Win,  Win16,  (int)D2D);
  f32_to_bf16_kernel<<<(int)(D2D / 256), 256, 0, stream>>>(Wout, Wout16, (int)D2D);

  const int STEPS = 8;   // matches reference setup; divisor read on device
  const dim3 g1(NTOK / 128, D_MODEL / 64);   // 32 x 16
  const dim3 g2(NTOK / 128, D_FFN / 64);     // 32 x 32
  for (int step = 0; step < STEPS; ++step) {
    const int sid = step < 8 ? step : 8;     // min(step, T_EMB-1)
    const float* temb = t_table + (size_t)sid * D_MODEL;

    ln_kernel<<<NTOK, 256, 0, stream>>>(h, nullptr, temb, attn_g, attn_b, x16);
    gemm_bf16_wmma_kernel<<<g1, 256, 0, stream>>>(x16, Wq16, nullptr, nullptr, q16, D_MODEL, D_MODEL, GM_BF16);
    gemm_bf16_wmma_kernel<<<g1, 256, 0, stream>>>(x16, Wk16, nullptr, nullptr, k16, D_MODEL, D_MODEL, GM_BF16);
    gemm_bf16_wmma_kernel<<<g1, 256, 0, stream>>>(x16, Wv16, nullptr, nullptr, v16, D_MODEL, D_MODEL, GM_BF16);
    transpose_v_kernel<<<(int)(TD / 256), 256, 0, stream>>>(v16, vT16);
    attention_kernel<<<1024, 256, 0, stream>>>(q16, k16, vT16, attn16);
    gemm_bf16_wmma_kernel<<<g1, 256, 0, stream>>>(attn16, Wo16, nullptr, attn_upd, nullptr, D_MODEL, D_MODEL, GM_F32);
    ln_kernel<<<NTOK, 256, 0, stream>>>(h, attn_upd, temb, ffn_g, ffn_b, x16);
    gemm_bf16_wmma_kernel<<<g2, 256, 0, stream>>>(x16, Win16, b_in, nullptr, hid16, D_MODEL, D_FFN, GM_SILU_BF16);
    gemm_bf16_wmma_kernel<<<g1, 256, 0, stream>>>(hid16, Wout16, b_out, ffn_upd, nullptr, D_FFN, D_MODEL, GM_BIAS_F32);
    update_kernel<<<NTOK, 256, 0, stream>>>(h, attn_upd, ffn_upd, steps_p);
  }
}
